// EncoderRNN_26920855011914
// MI455X (gfx1250) — matrix operand order repaired and checked by reading, not compiled, not run on hardware
//
#include <hip/hip_runtime.h>
#include <hip/hip_bf16.h>
#include <cstdint>

// Problem constants (B=128, S=512, I=128, Hd=512)
#define BB   128
#define SS   512
#define II   128
#define HD   512
#define MTOT (BB * SS)          // 65536 rows
#define GSZ  (3 * HD)           // 1536 gate columns

typedef __attribute__((ext_vector_type(16))) _Float16 v16h;
typedef __attribute__((ext_vector_type(8)))  _Float16 v8h;
typedef __attribute__((ext_vector_type(8)))  float    v8f;
typedef __attribute__((ext_vector_type(4)))  unsigned int u32x4;
typedef __attribute__((ext_vector_type(8)))  int          i32x8;
typedef __attribute__((ext_vector_type(4)))  int          i32x4;

union V16H { v16h v; v8h h[2]; };

// ---------------------------------------------------------------------------
// fp32 -> fp16 conversion for x, w_emb, w_ih
// ---------------------------------------------------------------------------
__global__ void cvt_f32_f16(const float* __restrict__ in, _Float16* __restrict__ out, int n) {
    int i = blockIdx.x * blockDim.x + threadIdx.x;
    if (i < n) out[i] = (_Float16)in[i];
}

// ---------------------------------------------------------------------------
// LDS geometry (halves unless noted)
// ---------------------------------------------------------------------------
#define ROWS_WG    128
#define LDS_STRID  520                       // 512 + 8 pad halves (1040 B rows)
#define EMB_H      (ROWS_WG * LDS_STRID)     // 66,560 halves  (133,120 B)
#define GATE_H     (16 * LDS_STRID)          // one gate tile: 16 padded rows
#define TILE_H     (3 * GATE_H)              // 24,960 halves  (49,920 B)
#define LDS_TOTAL_B ((EMB_H + 2 * TILE_H) * 2)   // 232,960 B  (< 320 KB)

// ---------------------------------------------------------------------------
// TDM: 1-D copy of one contiguous 16 KB gate block (16 rows x 1024 B) of
// w_ih(f16) into LDS, with pad_enable inserting 16 B after every 1024 B so
// the LDS row stride becomes 1040 B (bank-conflict-free column reads).
// Descriptor packing per CDNA5 ISA (D# groups 0/1).
// This toolchain exposes the 6-arg clang-23 builtin:
//   (u32x4 g0, i32x8 g1, i32x4 g2, i32x4 g3, i32x8 extra, i32 cpol)
// Groups 2/3 (and the extra group) are unused for <=2-D tiles -> zeros.
// ---------------------------------------------------------------------------
__device__ __forceinline__ void tdm_issue_gate(const _Float16* src, unsigned int lds_byte) {
    const unsigned long long ga = (unsigned long long)(uintptr_t)src;
    u32x4 g0;
    g0[0] = 1u;                                              // count=1, user desc
    g0[1] = lds_byte;                                        // LDS byte address
    g0[2] = (unsigned int)ga;                                // global_addr[31:0]
    g0[3] = ((unsigned int)(ga >> 32) & 0x01FFFFFFu)         // global_addr[56:32]
            | 0x80000000u;                                   // type=2 ("image")
    i32x8 g1;
    g1[0] = (int)((3u << 16)      // data_size = 8 B units
                | (1u << 20)      // pad_enable
                | (7u << 22)      // pad_interval: every 256 DW (1024 B)
                | (3u << 25));    // pad_amount: 4 DW (16 B)
    g1[1] = (int)(2048u << 16);   // tensor_dim0[15:0]=2048 (16 KB / 8 B)
    g1[2] = (int)(1u << 16);      // tensor_dim0[31:16]=0, tensor_dim1=1
    g1[3] = (int)(2048u << 16);   // tensor_dim1[31:16]=0, tile_dim0=2048
    g1[4] = 0;                    // tile_dim1=0 (unused), tile_dim2=0
    g1[5] = 2048;                 // tensor_dim0_stride[31:0]
    g1[6] = 0;                    // stride0[47:32], stride1[15:0]
    g1[7] = 0;                    // stride1[47:16]
    const i32x4 gz4 = {0, 0, 0, 0};                          // groups 2/3 unused
    const i32x8 gz8 = {0, 0, 0, 0, 0, 0, 0, 0};              // extra group unused
    __builtin_amdgcn_tensor_load_to_lds(g0, g1, gz4, gz4, gz8, 0);
}

__device__ __forceinline__ void tdm_issue_tile(const _Float16* wihh, int j, int buf) {
    const unsigned int base = (unsigned int)(EMB_H * 2) + (unsigned int)buf * (TILE_H * 2);
#pragma unroll
    for (int g = 0; g < 3; ++g) {
        const _Float16* src = wihh + (size_t)(g * HD + j * 16) * HD;   // 16 contiguous rows
        tdm_issue_gate(src, base + (unsigned int)g * (GATE_H * 2));
    }
}

// ---------------------------------------------------------------------------
// Fused: emb = x @ w_emb^T + b_emb (kept in LDS as f16)
//        gi  = emb @ w_ih^T + b_ih ; GRU-style gates ; scatter to (H, h_last)
// Grid: MTOT/128 workgroups, 256 threads (8 waves). Wave w owns rows
// [wg*128 + w*16, +16). Stage-2 B tiles staged via TDM, double-buffered,
// shared by all 8 waves (8x cut in L2 traffic).
// ---------------------------------------------------------------------------
__global__ void __launch_bounds__(256)
gru_fused_wmma(const _Float16* __restrict__ xh,     // [MTOT][II] f16
               const _Float16* __restrict__ wembh,  // [HD][II]   f16 (col-major B)
               const _Float16* __restrict__ wihh,   // [GSZ][HD]  f16 (col-major B)
               const float* __restrict__ b_emb,     // [HD]
               const float* __restrict__ b_ih,      // [GSZ]
               const float* __restrict__ b_hh,      // [GSZ]
               float* __restrict__ out)             // H (128*511*512) ++ h_last (128*512)
{
    extern __shared__ _Float16 lds[];               // EMB_H + 2*TILE_H halves

    const int lane   = threadIdx.x & 31;
    const int wave   = threadIdx.x >> 5;
    const int r0     = blockIdx.x * ROWS_WG;
    const int mbase  = wave * 16;                   // this wave's M-tile inside the WG
    const bool hiH   = lane >= 16;
    const int rowA   = lane & 15;                   // A-matrix row within tile
    const int kA     = hiH ? 16 : 0;                // A per-lane K sub-offset
    const int kB     = hiH ? 16 : 0;                // B per-lane K sub-offset
    const int colB   = lane & 15;                   // B/C/D column within tile

    // Kick off the DMA for gate tile 0 before stage 1 (hides TDM latency).
    if (wave == 0) tdm_issue_tile(wihh, 0, 0);

    // ------------------- Stage 1: emb tile (16 x 512) into LDS -------------
    {
        const _Float16* xrow = xh + (size_t)(r0 + mbase + rowA) * II;
        for (int j = 0; j < HD / 16; ++j) {         // 32 N-tiles
            const int n0 = j * 16;
            const _Float16* wcol = wembh + (size_t)(n0 + colB) * II;
            v8f acc = {};
#pragma unroll
            for (int kt = 0; kt < II / 32; ++kt) {  // 4 K-tiles
                const int k0 = kt * 32;
                V16H a, b;
                a.h[0] = *(const v8h*)(xrow + k0 + kA);        // K = k0+kA   .. +7
                a.h[1] = *(const v8h*)(xrow + k0 + kA + 8);    // K = k0+kA+8 .. +15
                b.v    = *(const v16h*)(wcol + k0 + kB);       // K = k0+kB   .. +15
                __builtin_prefetch(xrow + k0 + 64, 0, 1);      // global_prefetch_b8
                acc = __builtin_amdgcn_wmma_f32_16x16x32_f16(
                          false, a.v, false, b.v, (short)0, acc, false, false);
            }
            const float bias = b_emb[n0 + colB];
#pragma unroll
            for (int r = 0; r < 8; ++r) {
                const int m = mbase + r + (hiH ? 8 : 0);
                lds[m * LDS_STRID + n0 + colB] = (_Float16)(acc[r] + bias);
            }
        }
    }

    // ------------------- Stage 2: gi = emb @ w_ih^T, gates, scatter --------
    const _Float16* arow = lds + (size_t)(mbase + rowA) * LDS_STRID;
    for (int j = 0; j < HD / 16; ++j) {             // 32 gate-column tiles
        const int buf = j & 1;
        if (wave == 0) {
            if (j + 1 < HD / 16) {                  // prefetch next tile into other buffer
                tdm_issue_tile(wihh, j + 1, (j + 1) & 1);
                __builtin_amdgcn_s_wait_tensorcnt(3);   // tile j's 3 TDM ops done
            } else {
                __builtin_amdgcn_s_wait_tensorcnt(0);
            }
        }
        __syncthreads();                            // publish buf to all waves

        // B fragments from the padded LDS tile: column colB, K contiguous
        const _Float16* bt = lds + EMB_H + buf * TILE_H + colB * LDS_STRID;
        const _Float16* bR = bt;
        const _Float16* bZ = bt + GATE_H;
        const _Float16* bN = bt + 2 * GATE_H;

        v8f accR = {}, accZ = {}, accN = {};
#pragma unroll 4
        for (int kt = 0; kt < HD / 32; ++kt) {      // 16 K-tiles
            const int k0 = kt * 32;
            V16H a, fR, fZ, fN;
            a.h[0] = *(const v8h*)(arow + k0 + kA);            // ds_load_b128
            a.h[1] = *(const v8h*)(arow + k0 + kA + 8);
            fR.h[0] = *(const v8h*)(bR + k0 + kB);
            fR.h[1] = *(const v8h*)(bR + k0 + kB + 8);
            fZ.h[0] = *(const v8h*)(bZ + k0 + kB);
            fZ.h[1] = *(const v8h*)(bZ + k0 + kB + 8);
            fN.h[0] = *(const v8h*)(bN + k0 + kB);
            fN.h[1] = *(const v8h*)(bN + k0 + kB + 8);
            accR = __builtin_amdgcn_wmma_f32_16x16x32_f16(
                       false, a.v, false, fR.v, (short)0, accR, false, false);
            accZ = __builtin_amdgcn_wmma_f32_16x16x32_f16(
                       false, a.v, false, fZ.v, (short)0, accZ, false, false);
            accN = __builtin_amdgcn_wmma_f32_16x16x32_f16(
                       false, a.v, false, fN.v, (short)0, accN, false, false);
        }
        __syncthreads();                            // all LDS reads of buf done

        const int col = j * 16 + colB;
        const float bihR = b_ih[col],          bhhR = b_hh[col];
        const float bihZ = b_ih[HD + col],     bhhZ = b_hh[HD + col];
        const float bihN = b_ih[2 * HD + col], bhhN = b_hh[2 * HD + col];
#pragma unroll
        for (int r = 0; r < 8; ++r) {
            const int m  = mbase + r + (hiH ? 8 : 0);
            const int gr = r0 + m;                  // global row in [0, MTOT)
            const float ir  = accR[r] + bihR + bhhR;
            const float iz  = accZ[r] + bihZ + bhhZ;
            const float inn = accN[r] + bihN;
            const float rg  = 1.0f / (1.0f + __expf(-ir));
            const float zg  = 1.0f / (1.0f + __expf(-iz));
            const float ng  = tanhf(inn + rg * bhhN);
            const float hv  = (1.0f - zg) * ng;
            const int b = gr >> 9;                  // gr / SS
            const int s = gr & (SS - 1);            // gr % SS
            const size_t oidx = (s < SS - 1)
                ? ((size_t)b * (SS - 1) * HD + (size_t)s * HD + col)
                : ((size_t)BB * (SS - 1) * HD + (size_t)b * HD + col);
            out[oidx] = hv;
        }
    }
}

// ---------------------------------------------------------------------------
extern "C" void kernel_launch(void* const* d_in, const int* in_sizes, int n_in,
                              void* d_out, int out_size, void* d_ws, size_t ws_size,
                              hipStream_t stream) {
    const float* x     = (const float*)d_in[0];   // [128,512,128]
    const float* w_emb = (const float*)d_in[1];   // [512,128]
    const float* b_emb = (const float*)d_in[2];   // [512]
    const float* w_ih  = (const float*)d_in[3];   // [1536,512]
    const float* b_ih  = (const float*)d_in[4];   // [1536]
    const float* b_hh  = (const float*)d_in[5];   // [1536]
    float* out = (float*)d_out;

    // workspace: f16 copies of x, w_emb, w_ih
    const size_t nX  = (size_t)MTOT * II;         // 8,388,608
    const size_t nWE = (size_t)HD * II;           //    65,536
    const size_t nWI = (size_t)GSZ * HD;          //   786,432
    _Float16* xh    = (_Float16*)d_ws;
    _Float16* wembh = xh + nX;
    _Float16* wihh  = wembh + nWE;

    cvt_f32_f16<<<(int)((nX  + 255) / 256), 256, 0, stream>>>(x,     xh,    (int)nX);
    cvt_f32_f16<<<(int)((nWE + 255) / 256), 256, 0, stream>>>(w_emb, wembh, (int)nWE);
    cvt_f32_f16<<<(int)((nWI + 255) / 256), 256, 0, stream>>>(w_ih,  wihh,  (int)nWI);

    const int nWG = MTOT / ROWS_WG;                              // 512
    gru_fused_wmma<<<nWG, 256, LDS_TOTAL_B, stream>>>(
        xh, wembh, wihh, b_emb, b_ih, b_hh, out);
}
